// NeuroMUSX_V2_77068893159478
// MI455X (gfx1250) — matrix operand order repaired and checked by reading, not compiled, not run on hardware
//
#include <hip/hip_runtime.h>
#include <hip/hip_bf16.h>

#define GN 20000
#define GE 320000
#define GB 64

typedef __attribute__((ext_vector_type(16))) _Float16 v16h;
typedef __attribute__((ext_vector_type(8)))  _Float16 v8h;
typedef __attribute__((ext_vector_type(8)))  float    v8f;

// ---------------------------------------------------------------- utilities
__global__ void zero_f32(float* __restrict__ p, int n) {
    int i = blockIdx.x * blockDim.x + threadIdx.x;
    for (; i < n; i += gridDim.x * blockDim.x) p[i] = 0.0f;
}

// Convert two 128x128 f32 weight matrices to f16 (packed [2][128*128])
__global__ void conv_w16(const float* __restrict__ Wl, const float* __restrict__ Wr,
                         _Float16* __restrict__ w16) {
    int i = blockIdx.x * blockDim.x + threadIdx.x;   // 0..16383
    if (i < 128 * 128) {
        w16[i]             = (_Float16)Wl[i];
        w16[128 * 128 + i] = (_Float16)Wr[i];
    }
}

// ------------------------------------------------- WMMA GEMM (dual output):
//   XL = H(f16) @ Wl(f16),  XR = H(f16) @ Wr(f16)  — A fragments shared.
// H: [N,128] row-major f16.  W16: [2][128,128] row-major f16 (K x N).
// One wave -> one 16x16 tile of BOTH outputs. K=128 = 4 x WMMA(K=32) x 2 mats.
__global__ __launch_bounds__(32)
void wmma_proj128_dual(const _Float16* __restrict__ H16, const _Float16* __restrict__ W16,
                       float* __restrict__ XL, float* __restrict__ XR) {
    const int mt   = blockIdx.x;           // 0..1249  (M tile)
    const int nt   = blockIdx.y;           // 0..7     (N tile)
    const int lane = threadIdx.x;          // 0..31

    const _Float16* Wl = W16;
    const _Float16* Wr = W16 + 128 * 128;

    const int m0   = mt * 16;
    const int row  = m0 + (lane & 15);     // A: M = lane&15
    const int kgrp = (lane >> 4) * 8;      // A: lanes16-31 hold K-group +8

    v8f cl = {};
    v8f cr = {};
    #pragma unroll
    for (int k0 = 0; k0 < 128; k0 += 32) {
        // A fragment: lane holds row M, K = k0+kgrp+{0..7} and k0+kgrp+16+{0..7}
        const _Float16* ap = H16 + (size_t)row * 128 + k0 + kgrp;
        v8h lo = *(const v8h*)(ap);
        v8h hi = *(const v8h*)(ap + 16);
        v16h a;
        #pragma unroll
        for (int i = 0; i < 8; ++i) { a[i] = lo[i]; a[i + 8] = hi[i]; }
        // B fragments: lane holds col N = lane&15, K = k0+kgrp+{0..7} and k0+kgrp+16+{0..7}
        const size_t boff = (size_t)(k0 + kgrp) * 128 + nt * 16 + (lane & 15);
        v16h bl, br;
        #pragma unroll
        for (int i = 0; i < 8; ++i) {
            bl[i]     = Wl[boff + (size_t)i * 128];
            bl[i + 8] = Wl[boff + (size_t)(i + 16) * 128];
            br[i]     = Wr[boff + (size_t)i * 128];
            br[i + 8] = Wr[boff + (size_t)(i + 16) * 128];
        }
        cl = __builtin_amdgcn_wmma_f32_16x16x32_f16(false, a, false, bl,
                                                    (short)0, cl, false, false);
        cr = __builtin_amdgcn_wmma_f32_16x16x32_f16(false, a, false, br,
                                                    (short)0, cr, false, false);
    }
    // D layout: VGPR r -> M = 8*(lane>>4)+r, N = lane&15
    const int mrow = m0 + 8 * (lane >> 4);
    const int ncol = nt * 16 + (lane & 15);
    #pragma unroll
    for (int r = 0; r < 8; ++r) {
        XL[(size_t)(mrow + r) * 128 + ncol] = cl[r];
        XR[(size_t)(mrow + r) * 128 + ncol] = cr[r];
    }
}

// ------------------------------------------------- init layer projection (d_in=2)
__global__ void init_proj(const float* __restrict__ x, const float* __restrict__ Wl,
                          const float* __restrict__ Wr, float* __restrict__ xl,
                          float* __restrict__ xr) {
    int r = blockIdx.x, c = threadIdx.x;   // grid N x 128
    float x0 = x[2 * r], x1 = x[2 * r + 1];
    xl[(size_t)r * 128 + c] = x0 * Wl[c] + x1 * Wl[128 + c];
    xr[(size_t)r * 128 + c] = x0 * Wr[c] + x1 * Wr[128 + c];
}

// ------------------------------------------------- edge pass 1: logits -> exp, den
__global__ void edge_pass1(const int* __restrict__ src, const int* __restrict__ dst,
                           const float* __restrict__ ea2, const float* __restrict__ xl,
                           const float* __restrict__ xr, const float* __restrict__ We,
                           const float* __restrict__ att, float* __restrict__ ex,
                           float* __restrict__ den) {
    int t = blockIdx.x * blockDim.x + threadIdx.x;
    if (t >= GE * 8) return;
    int e = t >> 3, h = t & 7;
    int s = src[e], d = dst[e];
    float a0 = ea2[2 * e], a1 = ea2[2 * e + 1];
    const float* xls = xl + (size_t)s * 128 + h * 16;
    const float* xrd = xr + (size_t)d * 128 + h * 16;
    const float* w0  = We + h * 16;
    const float* w1  = We + 128 + h * 16;
    const float* at  = att + h * 16;
    float logit = 0.f;
    #pragma unroll
    for (int c = 0; c < 16; ++c) {
        float z = xls[c] + xrd[c] + a0 * w0[c] + a1 * w1[c];
        z = (z > 0.f) ? z : 0.2f * z;                 // leaky_relu 0.2
        logit += at[c] * z;
    }
    float ev = __expf(logit);                         // softmax w/o max-shift (equiv.)
    ex[t] = ev;
    atomicAdd(den + (size_t)d * 8 + h, ev);
}

// ------------------------------------------------- edge pass 2: weighted scatter
__global__ void edge_pass2(const int* __restrict__ src, const int* __restrict__ dst,
                           const float* __restrict__ xl, const float* __restrict__ ex,
                           const float* __restrict__ den, float* __restrict__ agg) {
    int e = blockIdx.x;                    // one block (128 thr) per edge
    int t = threadIdx.x;                   // channel 0..127, head = t>>4
    int s = src[e], d = dst[e];
    int h = t >> 4;
    float alpha = ex[(size_t)e * 8 + h] / (den[(size_t)d * 8 + h] + 1e-16f);
    atomicAdd(agg + (size_t)d * 128 + t, alpha * xl[(size_t)s * 128 + t]);
}

// ------------------------------------------------- batchnorm statistics
__global__ void bn_stats(const float* __restrict__ agg, const float* __restrict__ bias,
                         float* __restrict__ sums) {
    int c = threadIdx.x;                   // 128 channels
    float s = 0.f, q = 0.f;
    for (int r = blockIdx.x; r < GN; r += gridDim.x) {
        float v = agg[(size_t)r * 128 + c] + bias[c];
        s += v; q += v * v;
    }
    atomicAdd(sums + c, s);
    atomicAdd(sums + 128 + c, q);
}

// ------------------------------------------------- batchnorm apply + leaky + f16
__global__ void bn_apply(const float* __restrict__ agg, const float* __restrict__ bias,
                         const float* __restrict__ sums, const float* __restrict__ gamma,
                         const float* __restrict__ beta, float* __restrict__ h32,
                         _Float16* __restrict__ h16) {
    int r = blockIdx.x, c = threadIdx.x;
    const float inv_n = 1.0f / (float)GN;
    float mu  = sums[c] * inv_n;
    float var = sums[128 + c] * inv_n - mu * mu;
    float v   = agg[(size_t)r * 128 + c] + bias[c];
    float y   = gamma[c] * (v - mu) * rsqrtf(var + 1e-5f) + beta[c];
    y = (y > 0.f) ? y : 0.01f * y;                    // leaky_relu 0.01
    h32[(size_t)r * 128 + c] = y;
    h16[(size_t)r * 128 + c] = (_Float16)y;
}

// ------------------------------------------------- head-layer projection [128 -> 8]
__global__ void proj8(const float* __restrict__ h32, const float* __restrict__ Wl,
                      const float* __restrict__ Wr, float* __restrict__ xl8,
                      float* __restrict__ xr8) {
    int t = blockIdx.x * blockDim.x + threadIdx.x;
    if (t >= GN * 8) return;
    int r = t >> 3, hh = t & 7;
    const float* hr = h32 + (size_t)r * 128;
    float sl = 0.f, sr = 0.f;
    #pragma unroll 4
    for (int k = 0; k < 128; ++k) {
        float hv = hr[k];
        sl += hv * Wl[k * 8 + hh];
        sr += hv * Wr[k * 8 + hh];
    }
    xl8[t] = sl; xr8[t] = sr;
}

// ------------------------------------------------- head-layer edge passes (c=1)
__global__ void edge1_c1(const int* __restrict__ src, const int* __restrict__ dst,
                         const float* __restrict__ ea2, const float* __restrict__ xl8,
                         const float* __restrict__ xr8, const float* __restrict__ We,
                         const float* __restrict__ att, float* __restrict__ ex8,
                         float* __restrict__ den8) {
    int t = blockIdx.x * blockDim.x + threadIdx.x;
    if (t >= GE * 8) return;
    int e = t >> 3, h = t & 7;
    int s = src[e], d = dst[e];
    float z = xl8[(size_t)s * 8 + h] + xr8[(size_t)d * 8 + h]
            + ea2[2 * e] * We[h] + ea2[2 * e + 1] * We[8 + h];
    z = (z > 0.f) ? z : 0.2f * z;
    float ev = __expf(att[h] * z);
    ex8[t] = ev;
    atomicAdd(den8 + (size_t)d * 8 + h, ev);
}

__global__ void edge2_c1(const int* __restrict__ src, const int* __restrict__ dst,
                         const float* __restrict__ xl8, const float* __restrict__ ex8,
                         const float* __restrict__ den8, float* __restrict__ agg8) {
    int t = blockIdx.x * blockDim.x + threadIdx.x;
    if (t >= GE * 8) return;
    int e = t >> 3, h = t & 7;
    int s = src[e], d = dst[e];
    float alpha = ex8[t] / (den8[(size_t)d * 8 + h] + 1e-16f);
    atomicAdd(agg8 + (size_t)d * 8 + h, alpha * xl8[(size_t)s * 8 + h]);
}

// ------------------------------------------------- finalize mus / sat
__global__ void finalize_mean(const float* __restrict__ agg8, const float* __restrict__ bias,
                              float* __restrict__ outv) {
    int r = blockIdx.x * blockDim.x + threadIdx.x;
    if (r >= GN) return;
    float s = 0.f;
    #pragma unroll
    for (int h = 0; h < 8; ++h) s += agg8[(size_t)r * 8 + h];
    outv[r] = s * 0.125f + bias[0];
}

__global__ void finalize_sat(const float* __restrict__ agg8, const float* __restrict__ bias,
                             const int* __restrict__ batch, float* __restrict__ gsum,
                             float* __restrict__ gcnt) {
    int r = blockIdx.x * blockDim.x + threadIdx.x;
    if (r >= GN) return;
    float s = 0.f;
    #pragma unroll
    for (int h = 0; h < 8; ++h) s += agg8[(size_t)r * 8 + h];
    float v = s * 0.125f + bias[0];
    int b = batch[r];
    atomicAdd(gsum + b, v);
    atomicAdd(gcnt + b, 1.0f);
}

__global__ void sat_div(const float* __restrict__ gsum, const float* __restrict__ gcnt,
                        float* __restrict__ outv) {
    int b = threadIdx.x;
    if (b < GB) outv[b] = gsum[b] / fmaxf(gcnt[b], 1.0f);
}

// ================================================================ driver
extern "C" void kernel_launch(void* const* d_in, const int* in_sizes, int n_in,
                              void* d_out, int out_size, void* d_ws, size_t ws_size,
                              hipStream_t stream) {
    // ---- inputs (JAX pytree flatten order: dict keys sorted, 'W' < lowercase) ----
    const float* x    = (const float*)d_in[0];          // [N,2]
    const int*   eidx = (const int*)d_in[1];            // [2,E]
    const float* ea2  = (const float*)d_in[2];          // [E,2]
    const int*   bat  = (const int*)d_in[3];            // [N]
    const int*   src  = eidx;
    const int*   dst  = eidx + GE;

    // params: bn_hidden[5]{beta,gamma} @4..13, bn_init{beta,gamma} @14..15,
    // hidden[i]{We,Wl,Wr,att,bias} @16+5i, init @41..45, mus @46..50, sat @51..55
    const float* bnh_beta[5];  const float* bnh_gamma[5];
    for (int i = 0; i < 5; ++i) { bnh_beta[i]  = (const float*)d_in[4 + 2 * i];
                                  bnh_gamma[i] = (const float*)d_in[5 + 2 * i]; }
    const float* bni_beta  = (const float*)d_in[14];
    const float* bni_gamma = (const float*)d_in[15];
    const float *hWe[5], *hWl[5], *hWr[5], *hatt[5], *hbias[5];
    for (int i = 0; i < 5; ++i) {
        hWe[i]   = (const float*)d_in[16 + 5 * i];
        hWl[i]   = (const float*)d_in[17 + 5 * i];
        hWr[i]   = (const float*)d_in[18 + 5 * i];
        hatt[i]  = (const float*)d_in[19 + 5 * i];
        hbias[i] = (const float*)d_in[20 + 5 * i];
    }
    const float* iWe   = (const float*)d_in[41];
    const float* iWl   = (const float*)d_in[42];
    const float* iWr   = (const float*)d_in[43];
    const float* iatt  = (const float*)d_in[44];
    const float* ibias = (const float*)d_in[45];
    const float* mWe = (const float*)d_in[46]; const float* mWl = (const float*)d_in[47];
    const float* mWr = (const float*)d_in[48]; const float* matt = (const float*)d_in[49];
    const float* mbias = (const float*)d_in[50];
    const float* sWe = (const float*)d_in[51]; const float* sWl = (const float*)d_in[52];
    const float* sWr = (const float*)d_in[53]; const float* satt = (const float*)d_in[54];
    const float* sbias = (const float*)d_in[55];

    // ---- workspace carve ----
    float* ws = (float*)d_ws;
    size_t o = 0;
    float*     h32  = ws + o; o += (size_t)GN * 128;
    _Float16*  h16  = (_Float16*)(ws + o); o += (size_t)GN * 64;   // N*128 halves
    float*     xl   = ws + o; o += (size_t)GN * 128;
    float*     xr   = ws + o; o += (size_t)GN * 128;
    float*     agg  = ws + o; o += (size_t)GN * 128;
    float*     ex   = ws + o; o += (size_t)GE * 8;                 // reused for ex8
    float*     den  = ws + o; o += (size_t)GN * 8;
    float*     sums = ws + o; o += 256;
    _Float16*  w16  = (_Float16*)(ws + o); o += 16384;             // 2*128*128 halves
    float*     xl8  = ws + o; o += (size_t)GN * 8;
    float*     xr8  = ws + o; o += (size_t)GN * 8;
    float*     agg8 = ws + o; o += (size_t)GN * 8;
    float*     den8 = ws + o; o += (size_t)GN * 8;
    float*     gsum = ws + o; o += 64;
    float*     gcnt = ws + o; o += 64;

    float* out_mus = (float*)d_out;            // [N]
    float* out_sat = (float*)d_out + GN;       // [B]

    const int Z = 256;
    const dim3 gGemm(GN / 16, 8);
    const int nEdgeT = (GE * 8 + Z - 1) / Z;

    // ================= init layer =================
    init_proj<<<GN, 128, 0, stream>>>(x, iWl, iWr, xl, xr);
    zero_f32<<<512, Z, 0, stream>>>(den, GN * 8);
    zero_f32<<<2048, Z, 0, stream>>>(agg, GN * 128);
    zero_f32<<<1, Z, 0, stream>>>(sums, 256);
    edge_pass1<<<nEdgeT, Z, 0, stream>>>(src, dst, ea2, xl, xr, iWe, iatt, ex, den);
    edge_pass2<<<GE, 128, 0, stream>>>(src, dst, xl, ex, den, agg);
    bn_stats<<<256, 128, 0, stream>>>(agg, ibias, sums);
    bn_apply<<<GN, 128, 0, stream>>>(agg, ibias, sums, bni_gamma, bni_beta, h32, h16);

    // ================= hidden layers =================
    for (int i = 0; i < 5; ++i) {
        conv_w16<<<(128 * 128 + Z - 1) / Z, Z, 0, stream>>>(hWl[i], hWr[i], w16);
        wmma_proj128_dual<<<gGemm, 32, 0, stream>>>(h16, w16, xl, xr);
        zero_f32<<<512, Z, 0, stream>>>(den, GN * 8);
        zero_f32<<<2048, Z, 0, stream>>>(agg, GN * 128);
        zero_f32<<<1, Z, 0, stream>>>(sums, 256);
        edge_pass1<<<nEdgeT, Z, 0, stream>>>(src, dst, ea2, xl, xr, hWe[i], hatt[i], ex, den);
        edge_pass2<<<GE, 128, 0, stream>>>(src, dst, xl, ex, den, agg);
        bn_stats<<<256, 128, 0, stream>>>(agg, hbias[i], sums);
        bn_apply<<<GN, 128, 0, stream>>>(agg, hbias[i], sums, bnh_gamma[i], bnh_beta[i],
                                         h32, h16);
    }

    // ================= mus head =================
    proj8<<<(GN * 8 + Z - 1) / Z, Z, 0, stream>>>(h32, mWl, mWr, xl8, xr8);
    zero_f32<<<512, Z, 0, stream>>>(den8, GN * 8);
    zero_f32<<<512, Z, 0, stream>>>(agg8, GN * 8);
    edge1_c1<<<nEdgeT, Z, 0, stream>>>(src, dst, ea2, xl8, xr8, mWe, matt, ex, den8);
    edge2_c1<<<nEdgeT, Z, 0, stream>>>(src, dst, xl8, ex, den8, agg8);
    finalize_mean<<<(GN + Z - 1) / Z, Z, 0, stream>>>(agg8, mbias, out_mus);

    // ================= sat head =================
    proj8<<<(GN * 8 + Z - 1) / Z, Z, 0, stream>>>(h32, sWl, sWr, xl8, xr8);
    zero_f32<<<512, Z, 0, stream>>>(den8, GN * 8);
    zero_f32<<<512, Z, 0, stream>>>(agg8, GN * 8);
    zero_f32<<<1, 128, 0, stream>>>(gsum, 64);
    zero_f32<<<1, 128, 0, stream>>>(gcnt, 64);
    edge1_c1<<<nEdgeT, Z, 0, stream>>>(src, dst, ea2, xl8, xr8, sWe, satt, ex, den8);
    edge2_c1<<<nEdgeT, Z, 0, stream>>>(src, dst, xl8, ex, den8, agg8);
    finalize_sat<<<(GN + Z - 1) / Z, Z, 0, stream>>>(agg8, sbias, bat, gsum, gcnt);
    sat_div<<<1, 64, 0, stream>>>(gsum, gcnt, out_sat);
}
